// VQMoE_49864570306941
// MI455X (gfx1250) — matrix operand order repaired and checked by reading, not compiled, not run on hardware
//
#include <hip/hip_runtime.h>

#define BQ 64
#define NQ 1024
#define CQ 256
#define ROWS_TOTAL (BQ * NQ)   // 65536
#define BLK_ROWS 256           // rows per workgroup (8 waves x 32 rows)
#define CHUNK_K 128            // codebook codes staged per LDS chunk
#define LDS_STRIDE 264         // 256 + 8 bf16 pad -> 528B row stride, conflict-free

typedef __bf16 bf16_t;
typedef bf16_t v16bf __attribute__((ext_vector_type(16)));
typedef bf16_t v8bf  __attribute__((ext_vector_type(8)));
typedef float  v8f   __attribute__((ext_vector_type(8)));

__device__ __forceinline__ unsigned short f2bf(float f) {
    unsigned u = __float_as_uint(f);
    u = u + 0x7FFFu + ((u >> 16) & 1u);   // round-to-nearest-even
    return (unsigned short)(u >> 16);
}

// Async global->LDS copy (CDNA5, ASYNCcnt-tracked). The low 32 bits of a
// generic LDS pointer are the wave-relative LDS byte address (ISA 10.2).
__device__ __forceinline__ void async_g2l_b128(void* lds, const void* g) {
    asm volatile("global_load_async_to_lds_b128 %0, %1, off"
                 :: "v"((unsigned)(uintptr_t)lds),
                    "v"((unsigned long long)(uintptr_t)g)
                 : "memory");
}
__device__ __forceinline__ void async_g2l_b32(void* lds, const void* g) {
    asm volatile("global_load_async_to_lds_b32 %0, %1, off"
                 :: "v"((unsigned)(uintptr_t)lds),
                    "v"((unsigned long long)(uintptr_t)g)
                 : "memory");
}
__device__ __forceinline__ void wait_async0() {
#if __has_builtin(__builtin_amdgcn_s_wait_asynccnt)
    __builtin_amdgcn_s_wait_asynccnt(0);
#else
    asm volatile("s_wait_asynccnt 0" ::: "memory");
#endif
}

// ---------------------------------------------------------------------------
// Kernel 1: convert the 4 codebooks to bf16 (concatenated) + per-code ||c||^2
// ---------------------------------------------------------------------------
__global__ __launch_bounds__(256) void cb_prep_kernel(
    const float* __restrict__ cb0, const float* __restrict__ cb1,
    const float* __restrict__ cb2, const float* __restrict__ cb3,
    unsigned short* __restrict__ cb_bf, float* __restrict__ cnorm) {
    __shared__ float red[256];
    const int row = blockIdx.x;
    const int c = threadIdx.x;
    const float* src;
    int lr;
    if (row < 256)       { src = cb0; lr = row; }
    else if (row < 768)  { src = cb1; lr = row - 256; }
    else if (row < 1792) { src = cb2; lr = row - 768; }
    else                 { src = cb3; lr = row - 1792; }
    const float v = src[(size_t)lr * CQ + c];
    cb_bf[(size_t)row * CQ + c] = f2bf(v);
    red[c] = v * v;
    __syncthreads();
    for (int off = 128; off; off >>= 1) {
        if (c < off) red[c] += red[c + off];
        __syncthreads();
    }
    if (c == 0) cnorm[row] = red[0];
}

// ---------------------------------------------------------------------------
// Kernel 2: WMMA argmin. One block = (batch b, 256-row tile). 8 waves,
// wave w owns rows [32w, 32w+32) as TWO resident A sets; each B tile loaded
// from LDS feeds two independent WMMAs (halves DS traffic per WMMA).
// ---------------------------------------------------------------------------
__global__ __launch_bounds__(256) void vq_argmin_kernel(
    const float* __restrict__ z_e, const int* __restrict__ expert_idx,
    const unsigned short* __restrict__ cb_bf, const float* __restrict__ cnorm,
    int* __restrict__ idx_out) {
    extern __shared__ __align__(16) char smem[];
    unsigned short* zT  = (unsigned short*)smem;          // 256 x 264 bf16
    unsigned short* cT  = zT + BLK_ROWS * LDS_STRIDE;     // 128 x 264 bf16
    float*          cnT = (float*)(cT + CHUNK_K * LDS_STRIDE); // 128 f32

    const int bid = blockIdx.x;
    const int b = bid >> 2;                 // 4 blocks per batch
    const int rowBase = (bid & 3) * BLK_ROWS;
    const int t = threadIdx.x;

    const int r = expert_idx[b];
    const int Kr = 256 << r;
    const int cbRow = (256 << r) - 256;     // 0, 256, 768, 1792

    // --- stage z tile: f32 global -> bf16 LDS (RNE conversion in VALU) ---
    const float* zsrc = z_e + (size_t)(b * NQ + rowBase) * CQ;
    for (int i = t; i < (BLK_ROWS * CQ) / 4; i += 256) {
        const int flat = i * 4;
        const int rw = flat >> 8;
        const int col = flat & 255;
        const float4 f = *(const float4*)(zsrc + (size_t)rw * CQ + col);
        uint2 pk;
        pk.x = (unsigned)f2bf(f.x) | ((unsigned)f2bf(f.y) << 16);
        pk.y = (unsigned)f2bf(f.z) | ((unsigned)f2bf(f.w) << 16);
        *(uint2*)(zT + rw * LDS_STRIDE + col) = pk;
    }
    __syncthreads();

    const int w = t >> 5;
    const int lane = t & 31;
    const int half = lane >> 4;    // 0: lanes 0-15, 1: lanes 16-31
    const int l16 = lane & 15;

    // --- Two resident A sets: rows 32w+0..15 and 32w+16..31 (ISA A layout) ---
    // lane half 0: elems 0..7 -> K+0..7,  elems 8..15 -> K+16..23
    // lane half 1: elems 0..7 -> K+8..15, elems 8..15 -> K+24..31
    const unsigned short* zrow0 = zT + (32 * w + l16) * LDS_STRIDE;
    const unsigned short* zrow1 = zrow0 + 16 * LDS_STRIDE;
    v16bf A0[8], A1[8];
#pragma unroll
    for (int ks = 0; ks < 8; ++ks) {
        const int o = ks * 32 + 8 * half;
        v8bf lo = *(const v8bf*)(zrow0 + o);
        v8bf hi = *(const v8bf*)(zrow0 + o + 16);
        A0[ks] = __builtin_shufflevector(lo, hi, 0, 1, 2, 3, 4, 5, 6, 7,
                                         8, 9, 10, 11, 12, 13, 14, 15);
        lo = *(const v8bf*)(zrow1 + o);
        hi = *(const v8bf*)(zrow1 + o + 16);
        A1[ks] = __builtin_shufflevector(lo, hi, 0, 1, 2, 3, 4, 5, 6, 7,
                                         8, 9, 10, 11, 12, 13, 14, 15);
    }

    float best0[8], best1[8];
    int bidx0[8], bidx1[8];
#pragma unroll
    for (int v = 0; v < 8; ++v) {
        best0[v] = 3.0e38f; bidx0[v] = 0;
        best1[v] = 3.0e38f; bidx1[v] = 0;
    }

    const unsigned short* cbsrc = cb_bf + (size_t)cbRow * CQ;
    const float* cnsrc = cnorm + cbRow;

    for (int kb = 0; kb < Kr; kb += CHUNK_K) {
        __syncthreads();   // previous chunk fully consumed by all waves
        // stage 128 codes via async global->LDS (ASYNCcnt), 16B granules
        for (int i = t; i < (CHUNK_K * CQ) / 8; i += 256) {
            const int flat = i * 8;
            const int rw = flat >> 8;
            const int col = flat & 255;
            async_g2l_b128(cT + rw * LDS_STRIDE + col,
                           cbsrc + (size_t)(kb + rw) * CQ + col);
        }
        if (t < CHUNK_K)
            async_g2l_b32(cnT + t, cnsrc + kb + t);
        if (kb + CHUNK_K < Kr)
            __builtin_prefetch(cbsrc + (size_t)(kb + CHUNK_K + (t & 127)) * CQ, 0, 1);
        wait_async0();
        __syncthreads();

        // preload the 8 cnorm values for this chunk (one per code sub-tile)
        float cnv[8];
#pragma unroll
        for (int ct = 0; ct < 8; ++ct) cnv[ct] = cnT[ct * 16 + l16];

        for (int ct = 0; ct < 8; ++ct) {
            // B operand: lane half 0 -> K 0..7,16..23 of window, half 1 -> K 8..15,24..31
            const unsigned short* crow = cT + (ct * 16 + l16) * LDS_STRIDE + 8 * half;
            v8f acc0 = {0.f, 0.f, 0.f, 0.f, 0.f, 0.f, 0.f, 0.f};
            v8f acc1 = {0.f, 0.f, 0.f, 0.f, 0.f, 0.f, 0.f, 0.f};
#pragma unroll
            for (int ks = 0; ks < 8; ++ks) {
                const unsigned short* p = crow + ks * 32;
                const v8bf lo = *(const v8bf*)p;
                const v8bf hi = *(const v8bf*)(p + 16);
                const v16bf Bm = __builtin_shufflevector(lo, hi, 0, 1, 2, 3, 4, 5, 6, 7,
                                                         8, 9, 10, 11, 12, 13, 14, 15);
                acc0 = __builtin_amdgcn_wmma_f32_16x16x32_bf16(
                    false, A0[ks], false, Bm, (short)0, acc0, false, false);
                acc1 = __builtin_amdgcn_wmma_f32_16x16x32_bf16(
                    false, A1[ks], false, Bm, (short)0, acc1, false, false);
            }
            const float cn = cnv[ct];
            const int ci = kb + ct * 16 + l16;
#pragma unroll
            for (int v = 0; v < 8; ++v) {
                const float d0 = fmaf(-2.0f, acc0[v], cn);   // ||c||^2 - 2 z.c
                if (d0 < best0[v]) { best0[v] = d0; bidx0[v] = ci; }
                const float d1 = fmaf(-2.0f, acc1[v], cn);
                if (d1 < best1[v]) { best1[v] = d1; bidx1[v] = ci; }
            }
        }
    }

    // --- reduce argmin across the 16 column-lanes of each half ---
    const int rowG0 = b * NQ + rowBase + 32 * w + 8 * half;
#pragma unroll
    for (int v = 0; v < 8; ++v) {
        float bv = best0[v];
        int iv = bidx0[v];
#pragma unroll
        for (int off = 8; off >= 1; off >>= 1) {
            const float ob = __shfl_xor(bv, off, 32);
            const int oi = __shfl_xor(iv, off, 32);
            if (ob < bv || (ob == bv && oi < iv)) { bv = ob; iv = oi; }
        }
        if (l16 == 0) idx_out[rowG0 + v] = iv;        // rows 32w + v + 8*half
        bv = best1[v];
        iv = bidx1[v];
#pragma unroll
        for (int off = 8; off >= 1; off >>= 1) {
            const float ob = __shfl_xor(bv, off, 32);
            const int oi = __shfl_xor(iv, off, 32);
            if (ob < bv || (ob == bv && oi < iv)) { bv = ob; iv = oi; }
        }
        if (l16 == 0) idx_out[rowG0 + 16 + v] = iv;   // rows 32w+16 + v + 8*half
    }
}

// ---------------------------------------------------------------------------
// Kernel 3: gather z_q in f32, write z_q_out + indices(float), per-row ssd
// ---------------------------------------------------------------------------
__global__ __launch_bounds__(256) void vq_gather_kernel(
    const float* __restrict__ z_e, const int* __restrict__ expert_idx,
    const float* __restrict__ cb0, const float* __restrict__ cb1,
    const float* __restrict__ cb2, const float* __restrict__ cb3,
    const int* __restrict__ idx_in, float* __restrict__ zq_out,
    float* __restrict__ idxf_out, float* __restrict__ partials) {
    __shared__ float red[256];
    const int row = blockIdx.x;
    const int c = threadIdx.x;
    const int b = row >> 10;
    const int r = expert_idx[b];
    const float* cb = (r == 0) ? cb0 : (r == 1) ? cb1 : (r == 2) ? cb2 : cb3;
    const int idx = idx_in[row];
    const float zq = cb[(size_t)idx * CQ + c];
    const float ze = z_e[(size_t)row * CQ + c];
    zq_out[(size_t)row * CQ + c] = zq;   // straight-through value == z_q
    const float d = zq - ze;
    red[c] = d * d;
    __syncthreads();
    for (int off = 128; off; off >>= 1) {
        if (c < off) red[c] += red[c + off];
        __syncthreads();
    }
    if (c == 0) {
        partials[row] = red[0];
        idxf_out[row] = (float)idx;
    }
}

// ---------------------------------------------------------------------------
// Kernel 4: deterministic final reduce of 65536 partials -> scaled vq_loss
// ---------------------------------------------------------------------------
__global__ __launch_bounds__(256) void vq_loss_kernel(
    const float* __restrict__ partials, float* __restrict__ loss_out) {
    __shared__ float red[256];
    const int t = threadIdx.x;
    float s = 0.f;
    for (int i = t; i < ROWS_TOTAL; i += 256) s += partials[i];
    red[t] = s;
    __syncthreads();
    for (int off = 128; off; off >>= 1) {
        if (t < off) red[t] += red[t + off];
        __syncthreads();
    }
    // vq_loss = 1.25 * S_total / (B*N*C*C) = S_total * 1.25 / 2^32
    if (t == 0) loss_out[0] = red[0] * (1.25f / 4294967296.0f);
}

extern "C" void kernel_launch(void* const* d_in, const int* in_sizes, int n_in,
                              void* d_out, int out_size, void* d_ws, size_t ws_size,
                              hipStream_t stream) {
    const float* z_e = (const float*)d_in[0];
    const int* eidx  = (const int*)d_in[1];
    const float* cb0 = (const float*)d_in[2];
    const float* cb1 = (const float*)d_in[3];
    const float* cb2 = (const float*)d_in[4];
    const float* cb3 = (const float*)d_in[5];

    // workspace layout
    char* ws = (char*)d_ws;
    unsigned short* cb_bf = (unsigned short*)ws;                       // 3840*256*2
    float* cnorm   = (float*)(ws + (size_t)3840 * 256 * 2);            // 3840*4
    int*   idxbuf  = (int*)(ws + (size_t)3840 * 256 * 2 + 3840 * 4);   // 65536*4
    float* partials = (float*)(ws + (size_t)3840 * 256 * 2 + 3840 * 4
                               + (size_t)ROWS_TOTAL * 4);              // 65536*4

    float* zq_out   = (float*)d_out;                      // [64,1024,256]
    float* idxf_out = zq_out + (size_t)BQ * NQ * CQ;      // [64,1024] as float
    float* loss_out = idxf_out + (size_t)BQ * NQ;         // scalar

    cb_prep_kernel<<<3840, 256, 0, stream>>>(cb0, cb1, cb2, cb3, cb_bf, cnorm);

    const size_t shmem = (size_t)(BLK_ROWS + CHUNK_K) * LDS_STRIDE * 2 + CHUNK_K * 4; // ~203 KB
    vq_argmin_kernel<<<ROWS_TOTAL / BLK_ROWS, 256, shmem, stream>>>(
        z_e, eidx, cb_bf, cnorm, idxbuf);

    vq_gather_kernel<<<ROWS_TOTAL, 256, 0, stream>>>(
        z_e, eidx, cb0, cb1, cb2, cb3, idxbuf, zq_out, idxf_out, partials);

    vq_loss_kernel<<<1, 256, 0, stream>>>(partials, loss_out);
}
